// lstm_40458591928585
// MI455X (gfx1250) — compile-verified, matrix operand order checked by reading, not run on hardware
//
#include <hip/hip_runtime.h>

// ---------------------------------------------------------------------------
// Persistent fused LSTM for MI455X (gfx1250, wave32, WMMA bf16), round 2.
//   B=512, S=1024, I=64, H=128, O=1
// 32 workgroups x 256 threads; each WG owns 16 batch rows for all 1024 steps.
// W_hh B-fragments are register-resident (loop-invariant); W_ih in LDS.
// Ping-pong x/h LDS buffers -> single barrier per timestep.
// ---------------------------------------------------------------------------

typedef __attribute__((ext_vector_type(16))) __bf16 v16bf;
typedef __attribute__((ext_vector_type(8)))  __bf16 v8bf;
typedef __attribute__((ext_vector_type(4)))  __bf16 v4bf;
typedef __attribute__((ext_vector_type(8)))  float  v8f;
typedef __attribute__((ext_vector_type(4)))  float  v4f;

#define BB   512
#define SS   1024
#define II   64
#define HH   128
#define G4   512        // 4*H
#define WP   136        // padded pitch (halfs) for H=128 rows (272B)
#define XP   72         // padded pitch (halfs) for I=64 rows  (144B)

// LDS layout in __bf16 units
#define OFF_WIH   0                        // 512*XP
#define OFF_XBUF  (512*XP)                 // 2 * 16*XP  (ping-pong)
#define OFF_HBUF  (OFF_XBUF + 2*16*XP)     // 2 * 16*WP  (ping-pong)
#define XBSZ      (16*XP)
#define HBSZ      (16*WP)
#define LDS_BYTES ((OFF_HBUF + 2*HBSZ)*2)

__device__ __forceinline__ v16bf load_frag(const __bf16* p) {
    // 16 halfs per lane: [k .. k+7] and [k+16 .. k+23]  (ISA 7.12.2 16-bit layout)
    v8bf lo = *(const v8bf*)(p);
    v8bf hi = *(const v8bf*)(p + 16);
    return __builtin_shufflevector(lo, hi, 0,1,2,3,4,5,6,7,8,9,10,11,12,13,14,15);
}

// Load one B-fragment worth of a weight row directly from global f32 -> bf16 regs.
__device__ __forceinline__ v16bf load_wfrag_global(const float* p) {
    v4f a = *(const v4f*)(p);
    v4f b = *(const v4f*)(p + 4);
    v4f c = *(const v4f*)(p + 16);
    v4f d = *(const v4f*)(p + 20);
    v16bf r;
    r[0]  = (__bf16)a[0]; r[1]  = (__bf16)a[1]; r[2]  = (__bf16)a[2]; r[3]  = (__bf16)a[3];
    r[4]  = (__bf16)b[0]; r[5]  = (__bf16)b[1]; r[6]  = (__bf16)b[2]; r[7]  = (__bf16)b[3];
    r[8]  = (__bf16)c[0]; r[9]  = (__bf16)c[1]; r[10] = (__bf16)c[2]; r[11] = (__bf16)c[3];
    r[12] = (__bf16)d[0]; r[13] = (__bf16)d[1]; r[14] = (__bf16)d[2]; r[15] = (__bf16)d[3];
    return r;
}

__device__ __forceinline__ v8f wmma_bf16(v16bf a, v16bf b, v8f c) {
    return __builtin_amdgcn_wmma_f32_16x16x32_bf16(
        /*neg_a=*/false, a, /*neg_b=*/false, b,
        /*c_mod=*/(short)0, c, /*reuse_a=*/false, /*reuse_b=*/false);
}

__device__ __forceinline__ float hw_tanh(float x) {
#if __has_builtin(__builtin_amdgcn_tanhf)
    return __builtin_amdgcn_tanhf(x);           // native V_TANH_F32 (CDNA5 TRANS op)
#elif __has_builtin(__builtin_amdgcn_tanh_f32)
    return __builtin_amdgcn_tanh_f32(x);
#else
    return 1.0f - 2.0f * __builtin_amdgcn_rcpf(__expf(2.0f * x) + 1.0f);
#endif
}
__device__ __forceinline__ float hw_sigmoid(float x) {
    return 0.5f + 0.5f * hw_tanh(0.5f * x);     // 1 trans op instead of exp+rcp
}

__global__ void __launch_bounds__(256, 1)
lstm_persistent_kernel(const float* __restrict__ x,
                       const float* __restrict__ Wih,
                       const float* __restrict__ Whh,
                       const float* __restrict__ bih,
                       const float* __restrict__ bhh,
                       const float* __restrict__ Wout,
                       const float* __restrict__ bout,
                       float* __restrict__ out)
{
    extern __shared__ __align__(16) char smem_raw[];
    __bf16* smem = (__bf16*)smem_raw;
    __bf16* wih  = smem + OFF_WIH;            // [512][XP] bf16
    __bf16* xbuf = smem + OFF_XBUF;           // 2 x [16][XP] bf16 (ping-pong)
    __bf16* hbuf = smem + OFF_HBUF;           // 2 x [16][WP] bf16 (ping-pong)

    const int tid = threadIdx.x;
    const int b0  = blockIdx.x * 16;          // this WG's batch rows

    // ---- stage W_ih (512x64 f32 -> bf16, padded pitch XP) ---------------------
    for (int i = tid; i < (G4 * II) / 4; i += 256) {      // 8192 float4
        int row = i >> 4;
        int c4  = (i & 15) << 2;
        v4f v = *(const v4f*)(Wih + row * II + c4);
        v4bf b; b[0] = (__bf16)v[0]; b[1] = (__bf16)v[1];
                b[2] = (__bf16)v[2]; b[3] = (__bf16)v[3];
        *(v4bf*)(wih + row * XP + c4) = b;
    }
    // ---- h0 = 0 (only buffer 0 is read at t=0) --------------------------------
    for (int j = tid; j < HBSZ; j += 256) hbuf[j] = (__bf16)0.0f;

    // ---- per-wave constants ---------------------------------------------------
    const int lane = tid & 31;
    const int wv   = tid >> 5;                 // wave 0..7
    const int col  = lane & 15;                // N within tile / A-row M
    const int hid  = wv * 16 + col;            // hidden unit owned by this lane
    const int ksel = (lane >> 4) << 3;         // 0 (lanes 0-15) or 8 (lanes 16-31)
    const int rr[4] = { hid, HH + hid, 2 * HH + hid, 3 * HH + hid };

    // Biases (column-broadcast), read straight from global
    float bias[4];
    #pragma unroll
    for (int g = 0; g < 4; ++g) bias[g] = bih[rr[g]] + bhh[rr[g]];

    // ---- W_hh B-fragments: loop-invariant -> keep in VGPRs (16 x 8 = 128 regs)
    v16bf wh[4][4];
    #pragma unroll
    for (int g = 0; g < 4; ++g)
        #pragma unroll
        for (int k = 0; k < 4; ++k)
            wh[g][k] = load_wfrag_global(Whh + (size_t)rr[g] * HH + k * 32 + ksel);

    v8f cst;                                   // cell state, C-layout fragment
    #pragma unroll
    for (int v = 0; v < 8; ++v) cst[v] = 0.0f;

    __syncthreads();                           // wih + h0 staged

    // ---- time loop: ONE barrier per step (ping-pong buffers) ------------------
    for (int t = 0; t < SS; ++t) {
        __bf16* xb = xbuf + (t & 1) * XBSZ;
        const __bf16* hr = hbuf + (t & 1) * HBSZ;          // read h_{t-1}
        __bf16* hw = hbuf + ((t + 1) & 1) * HBSZ;          // write h_t

        // stage x_t tile [16 x 64] f32 -> bf16 LDS (256 threads, 4 elems each)
        {
            int row = tid >> 4;
            int c4  = (tid & 15) << 2;
            v4f v = *(const v4f*)(x + ((size_t)(b0 + row) * SS + t) * II + c4);
            v4bf b; b[0] = (__bf16)v[0]; b[1] = (__bf16)v[1];
                    b[2] = (__bf16)v[2]; b[3] = (__bf16)v[3];
            *(v4bf*)(xb + row * XP + c4) = b;
        }
        // prefetch next timestep's x tile (global_prefetch_b8)
        if (t + 1 < SS && tid < 32) {
            int pr = tid >> 1, part = (tid & 1) << 5;
            __builtin_prefetch(x + ((size_t)(b0 + pr) * SS + (t + 1)) * II + part, 0, 1);
        }
        __syncthreads();   // x_t staged; h_{t-1} writes visible; prev reads done

        // accumulators = bias (broadcast per gate column)
        v8f ai, af, ag, ao;
        #pragma unroll
        for (int v = 0; v < 8; ++v) {
            ai[v] = bias[0]; af[v] = bias[1]; ag[v] = bias[2]; ao[v] = bias[3];
        }

        // x_t @ W_ih^T : K = 64 (weights from LDS)
        #pragma unroll
        for (int k = 0; k < 2; ++k) {
            v16bf a = load_frag(xb + col * XP + k * 32 + ksel);
            ai = wmma_bf16(a, load_frag(wih + rr[0] * XP + k * 32 + ksel), ai);
            af = wmma_bf16(a, load_frag(wih + rr[1] * XP + k * 32 + ksel), af);
            ag = wmma_bf16(a, load_frag(wih + rr[2] * XP + k * 32 + ksel), ag);
            ao = wmma_bf16(a, load_frag(wih + rr[3] * XP + k * 32 + ksel), ao);
        }
        // h_{t-1} @ W_hh^T : K = 128 (weights register-resident, no waits)
        #pragma unroll
        for (int k = 0; k < 4; ++k) {
            v16bf a = load_frag(hr + col * WP + k * 32 + ksel);
            ai = wmma_bf16(a, wh[0][k], ai);
            af = wmma_bf16(a, wh[1][k], af);
            ag = wmma_bf16(a, wh[2][k], ag);
            ao = wmma_bf16(a, wh[3][k], ao);
        }

        // gates -> new cell & hidden; write h_t into the other buffer
        #pragma unroll
        for (int v = 0; v < 8; ++v) {
            float iv = hw_sigmoid(ai[v]);
            float fv = hw_sigmoid(af[v]);
            float gv = hw_tanh(ag[v]);
            float ov = hw_sigmoid(ao[v]);
            float cs = fv * cst[v] + iv * gv;
            cst[v] = cs;
            float hv = ov * hw_tanh(cs);
            int mr = v + ksel;                 // C-layout: M = v (+8 for hi lanes)
            hw[mr * WP + hid] = (__bf16)hv;
        }
    }
    __syncthreads();

    // ---- output projection: out[b] = h_n[b,:] . W_out + b_out  (O = 1) --------
    // Final h was written to buffer ((SS-1)+1)&1 == 0.
    if (tid < 16) {
        float s = bout[0];
        const __bf16* hrow = hbuf + tid * WP;
        #pragma unroll 8
        for (int n = 0; n < HH; ++n) s += (float)hrow[n] * Wout[n];
        out[b0 + tid] = s;
    }
}

extern "C" void kernel_launch(void* const* d_in, const int* in_sizes, int n_in,
                              void* d_out, int out_size, void* d_ws, size_t ws_size,
                              hipStream_t stream) {
    (void)in_sizes; (void)n_in; (void)out_size; (void)d_ws; (void)ws_size;
    const float* x    = (const float*)d_in[0];
    const float* Wih  = (const float*)d_in[1];
    const float* Whh  = (const float*)d_in[2];
    const float* bih  = (const float*)d_in[3];
    const float* bhh  = (const float*)d_in[4];
    const float* Wout = (const float*)d_in[5];
    const float* bout = (const float*)d_in[6];
    float* out = (float*)d_out;

    (void)hipFuncSetAttribute((const void*)lstm_persistent_kernel,
                              hipFuncAttributeMaxDynamicSharedMemorySize,
                              (int)LDS_BYTES);

    lstm_persistent_kernel<<<BB / 16, 256, LDS_BYTES, stream>>>(
        x, Wih, Whh, bih, bhh, Wout, bout, out);
}
